// XSimGCL_8461085573263
// MI455X (gfx1250) — matrix operand order repaired and checked by reading, not compiled, not run on hardware
//
#include <hip/hip_runtime.h>
#include <hip/hip_bf16.h>

// ---------------- problem constants (from reference) ----------------
#define UCNT 50000
#define ICNT 25000
#define NCNT 75000           // U + I
#define DIM  64
#define BSZ  4096
#define EPSV 0.2f
#define TEMPV 0.2f
#define REG_LAM 1.0e-4f
#define SSL_LAM 0.2f

typedef __attribute__((ext_vector_type(16))) _Float16 v16h;
typedef __attribute__((ext_vector_type(8)))  float    v8f;

// ---------------- small utility kernels ----------------
__global__ void xs_zero(float* __restrict__ p, long long n) {
  long long i = (long long)blockIdx.x * blockDim.x + threadIdx.x;
  if (i < n) p[i] = 0.f;
}

__global__ void xs_concat(const float* __restrict__ ut, const float* __restrict__ it,
                          float* __restrict__ x) {
  int i = blockIdx.x * blockDim.x + threadIdx.x;
  const int total = NCNT * DIM;
  if (i >= total) return;
  const int ud = UCNT * DIM;
  x[i] = (i < ud) ? ut[i] : it[i - ud];
}

// ---------------- SpMM: segment_sum(edge_val * x[dst] -> src) ----------------
// L2-resident (x is 19.2MB << 192MB L2); scatter via global_atomic_add_f32.
__global__ void xs_spmm(const int* __restrict__ src, const int* __restrict__ dst,
                        const float* __restrict__ val,
                        const float* __restrict__ xin, float* __restrict__ xout,
                        int e2) {
  long long gid = (long long)blockIdx.x * blockDim.x + threadIdx.x;
  long long total = (long long)e2 * DIM;
  if (gid >= total) return;
  int e = (int)(gid >> 6);
  int d = (int)(gid & 63);
  float v = val[e] * xin[(long long)dst[e] * DIM + d];
  atomicAdd(&xout[(long long)src[e] * DIM + d], v);
}

// ---------------- per-row noise injection + running layer-mean ----------------
// x[n] += sign(x[n]) * l2norm(noise[n]) * EPS ; emb += x/3.  One wave32 per row.
__global__ void xs_noise_mean(float* __restrict__ x, const float* __restrict__ noise,
                              float* __restrict__ emb) {
  int wave = (blockIdx.x * blockDim.x + threadIdx.x) >> 5;
  int lane = threadIdx.x & 31;
  if (wave >= NCNT) return;
  const float* nr = noise + (long long)wave * DIM;
  float a = nr[lane], b = nr[lane + 32];
  float ss = a * a + b * b;
  #pragma unroll
  for (int m = 16; m >= 1; m >>= 1) ss += __shfl_xor(ss, m, 32);
  float inv = 1.f / fmaxf(sqrtf(ss), 1e-12f);
  float* xr = x + (long long)wave * DIM;
  float* er = emb + (long long)wave * DIM;
  #pragma unroll
  for (int j = 0; j < 2; ++j) {
    int d = lane + 32 * j;
    float xv = xr[d];
    float sg = (xv > 0.f) ? 1.f : ((xv < 0.f) ? -1.f : 0.f);
    xv += sg * (nr[d] * inv) * EPSV;
    xr[d] = xv;
    er[d] += xv * (1.f / 3.f);
  }
}

// ---------------- batched row gather with scale ----------------
__global__ void xs_gather(const float* __restrict__ x, const int* __restrict__ idx,
                          int offset, float scale, float* __restrict__ out) {
  int i = blockIdx.x * blockDim.x + threadIdx.x;  // b*64 + d
  if (i >= BSZ * DIM) return;
  int b = i >> 6, d = i & 63;
  out[i] = x[(long long)(idx[b] + offset) * DIM + d] * scale;
}

// ---------------- BPR + reg losses: one wave per batch row ----------------
__global__ void xs_bpr_reg(const float* __restrict__ ue, const float* __restrict__ pe,
                           const float* __restrict__ ne,
                           const float* __restrict__ ut, const float* __restrict__ it,
                           const int* __restrict__ user, const int* __restrict__ pos,
                           const int* __restrict__ neg, float* __restrict__ out) {
  int wave = (blockIdx.x * blockDim.x + threadIdx.x) >> 5;
  int lane = threadIdx.x & 31;
  if (wave >= BSZ) return;
  float ps = 0.f, ns = 0.f, rg = 0.f;
  #pragma unroll
  for (int j = 0; j < 2; ++j) {
    int d = lane + 32 * j;
    float u = ue[wave * DIM + d], p = pe[wave * DIM + d], n = ne[wave * DIM + d];
    ps += u * p;
    ns += u * n;
    float eu = ut[(long long)user[wave] * DIM + d];
    float ep = it[(long long)pos[wave]  * DIM + d];
    float en = it[(long long)neg[wave]  * DIM + d];
    rg += eu * eu + ep * ep + en * en;
  }
  #pragma unroll
  for (int m = 16; m >= 1; m >>= 1) {
    ps += __shfl_xor(ps, m, 32);
    ns += __shfl_xor(ns, m, 32);
    rg += __shfl_xor(rg, m, 32);
  }
  if (lane == 0) {
    float z = ns - ps;
    float sp = (z > 20.f) ? z : log1pf(__expf(z));      // softplus
    atomicAdd(&out[0], sp * (1.f / BSZ));
    atomicAdd(&out[1], REG_LAM * 0.5f * rg * (1.f / BSZ));
  }
}

// ---------------- InfoNCE prep: row norms + positive scores ----------------
__global__ void xs_norms_pos(const float* __restrict__ v1, const float* __restrict__ v2,
                             float* __restrict__ n1, float* __restrict__ n2,
                             float* __restrict__ pos) {
  int wave = (blockIdx.x * blockDim.x + threadIdx.x) >> 5;
  int lane = threadIdx.x & 31;
  if (wave >= BSZ) return;
  float s1 = 0.f, s2 = 0.f, dp = 0.f;
  #pragma unroll
  for (int j = 0; j < 2; ++j) {
    int d = lane + 32 * j;
    float a = v1[wave * DIM + d];
    float b = v2[wave * DIM + d];
    s1 += a * a; s2 += b * b; dp += a * b;
  }
  #pragma unroll
  for (int m = 16; m >= 1; m >>= 1) {
    s1 += __shfl_xor(s1, m, 32);
    s2 += __shfl_xor(s2, m, 32);
    dp += __shfl_xor(dp, m, 32);
  }
  if (lane == 0) {
    float a1 = fmaxf(sqrtf(s1), 1e-12f);
    float a2 = fmaxf(sqrtf(s2), 1e-12f);
    n1[wave] = a1;
    n2[wave] = a2;
    pos[wave] = dp / (a1 * a2 * TEMPV);
  }
}

// ---------------- fragment packing per ISA 7.12.2 (16-bit layouts, wave32) ----
// frag flat layout: [strip(256)][kfrag(2)][lane(32)][elem(16)] halves.
// A (16x32 MxK):  m = lane%16 ; v=e/2, lo=e%2, h=lane/16
//                 k_local = v<4 ? 2v+lo+8h : 16+2(v-4)+lo+8h
// B (32x16 KxN):  n = lane%16 ; k_local = e<8 ? e+8h : 16+(e-8)+8h
__global__ void xs_pack(const float* __restrict__ v, const float* __restrict__ nrm,
                        _Float16* __restrict__ frag, int is_b) {
  int t = blockIdx.x * blockDim.x + threadIdx.x;
  if (t >= 256 * 2 * 32 * 16) return;
  int e = t & 15;
  int l = (t >> 4) & 31;
  int f = (t >> 9) & 1;
  int s = t >> 10;
  int h = l >> 4;
  int row, k;
  if (!is_b) {
    int m  = l & 15;
    int vv = e >> 1, lo = e & 1;
    int kl = (vv < 4) ? (2 * vv + lo + 8 * h) : (16 + 2 * (vv - 4) + lo + 8 * h);
    row = s * 16 + m;
    k   = f * 32 + kl;
  } else {
    int n  = l & 15;
    int kl = (e < 8) ? (e + 8 * h) : (16 + (e - 8) + 8 * h);
    row = s * 16 + n;             // column of S == row of v2
    k   = f * 32 + kl;
  }
  frag[t] = (_Float16)(v[(long long)row * DIM + k] / nrm[row]);
}

// ---------------- WMMA GEMM + fused exp/row-sum (never materializes S) -------
// One wave per 16-row strip; 256 column tiles x 2 K-fragments = 512 WMMAs/wave.
__global__ void xs_gemm_lse(const _Float16* __restrict__ fragA,
                            const _Float16* __restrict__ fragB,
                            float* __restrict__ rowsum) {
  int wave = (blockIdx.x * blockDim.x + threadIdx.x) >> 5;  // strip 0..255
  int lane = threadIdx.x & 31;
  if (wave >= 256) return;
  const v16h* aB = (const v16h*)(fragA + (size_t)wave * 1024);
  v16h a0 = aB[lane];
  v16h a1 = aB[32 + lane];
  v8f acc = {0.f, 0.f, 0.f, 0.f, 0.f, 0.f, 0.f, 0.f};
  const float invT = 1.f / TEMPV;
  for (int ct = 0; ct < 256; ++ct) {
    const v16h* bB = (const v16h*)(fragB + (size_t)ct * 1024);
    v16h b0 = bB[lane];
    v16h b1 = bB[32 + lane];
    v8f c = {0.f, 0.f, 0.f, 0.f, 0.f, 0.f, 0.f, 0.f};
    c = __builtin_amdgcn_wmma_f32_16x16x32_f16(false, a0, false, b0,
                                               (short)0, c, false, false);
    c = __builtin_amdgcn_wmma_f32_16x16x32_f16(false, a1, false, b1,
                                               (short)0, c, false, false);
    // sim/TEMP in [-5,5] -> exp never overflows; skip max-subtraction.
    #pragma unroll
    for (int v = 0; v < 8; ++v) acc[v] += __expf(c[v] * invT);
  }
  // C layout: VGPR v, lanes 0-15 -> row (16w+v) cols 0-15; lanes 16-31 -> row (16w+v+8).
  #pragma unroll
  for (int v = 0; v < 8; ++v) {
    float s = acc[v];
    #pragma unroll
    for (int m = 8; m >= 1; m >>= 1) s += __shfl_xor(s, m, 16);
    if ((lane & 15) == 0) rowsum[wave * 16 + v + 8 * (lane >> 4)] = s;
  }
}

__global__ void xs_ssl_fin(const float* __restrict__ rowsum,
                           const float* __restrict__ pos, float* __restrict__ out) {
  int i = blockIdx.x * blockDim.x + threadIdx.x;
  if (i >= BSZ) return;
  atomicAdd(&out[2], SSL_LAM * (logf(rowsum[i]) - pos[i]) * (1.f / BSZ));
}

// ---------------- host orchestration ----------------
extern "C" void kernel_launch(void* const* d_in, const int* in_sizes, int n_in,
                              void* d_out, int out_size, void* d_ws, size_t ws_size,
                              hipStream_t stream) {
  const float* user_table = (const float*)d_in[0];
  const float* item_table = (const float*)d_in[1];
  const float* edge_val   = (const float*)d_in[2];
  const float* noise      = (const float*)d_in[3];
  const int*   edge_src   = (const int*)d_in[4];
  const int*   edge_dst   = (const int*)d_in[5];
  const int*   user       = (const int*)d_in[6];
  const int*   positive   = (const int*)d_in[7];
  const int*   negative   = (const int*)d_in[8];
  const int    e2         = in_sizes[2];
  float* out = (float*)d_out;

  // bump allocator on workspace
  size_t off = 0;
  char* base = (char*)d_ws;
  auto alloc = [&](size_t bytes) -> void* {
    void* p = base + off;
    off += (bytes + 255) & ~(size_t)255;
    return p;
  };
  const size_t ND = (size_t)NCNT * DIM;
  float* xa   = (float*)alloc(ND * 4);
  float* xb   = (float*)alloc(ND * 4);
  float* emb  = (float*)alloc(ND * 4);
  float* ucl  = (float*)alloc((size_t)BSZ * DIM * 4);
  float* icl  = (float*)alloc((size_t)BSZ * DIM * 4);
  float* ue   = (float*)alloc((size_t)BSZ * DIM * 4);
  float* pe   = (float*)alloc((size_t)BSZ * DIM * 4);
  float* ne   = (float*)alloc((size_t)BSZ * DIM * 4);
  float* n1   = (float*)alloc((size_t)BSZ * 4);
  float* n2   = (float*)alloc((size_t)BSZ * 4);
  float* poss = (float*)alloc((size_t)BSZ * 4);
  _Float16* fragA = (_Float16*)alloc((size_t)256 * 2 * 32 * 16 * 2);
  _Float16* fragB = (_Float16*)alloc((size_t)256 * 2 * 32 * 16 * 2);
  float* rowsum = (float*)alloc((size_t)BSZ * 4);

  const int TB = 256;
  auto blk = [](long long n, int tb) { return (unsigned)((n + tb - 1) / tb); };

  // zero loss outputs + running mean
  xs_zero<<<1, 32, 0, stream>>>(out, 3);
  xs_zero<<<blk(ND, TB), TB, 0, stream>>>(emb, (long long)ND);

  // x = concat(user_table, item_table)
  xs_concat<<<blk(ND, TB), TB, 0, stream>>>(user_table, item_table, xa);

  float* xcur = xa;
  float* xnext = xb;
  for (int layer = 0; layer < 3; ++layer) {
    xs_zero<<<blk(ND, TB), TB, 0, stream>>>(xnext, (long long)ND);
    xs_spmm<<<blk((long long)e2 * DIM, TB), TB, 0, stream>>>(
        edge_src, edge_dst, edge_val, xcur, xnext, e2);
    xs_noise_mean<<<blk((long long)NCNT * 32, TB), TB, 0, stream>>>(
        xnext, noise + (size_t)layer * ND, emb);
    if (layer == 0) {  // CL_LAYER-1
      xs_gather<<<blk(BSZ * DIM, TB), TB, 0, stream>>>(xnext, user, 0, 1.f, ucl);
      xs_gather<<<blk(BSZ * DIM, TB), TB, 0, stream>>>(xnext, positive, UCNT, 1.f, icl);
    }
    float* t = xcur; xcur = xnext; xnext = t;
  }

  // final = mean(embs); gather batch rows (emb already holds sum/3 via 1/3 accum)
  xs_gather<<<blk(BSZ * DIM, TB), TB, 0, stream>>>(emb, user, 0, 1.f, ue);
  xs_gather<<<blk(BSZ * DIM, TB), TB, 0, stream>>>(emb, positive, UCNT, 1.f, pe);
  xs_gather<<<blk(BSZ * DIM, TB), TB, 0, stream>>>(emb, negative, UCNT, 1.f, ne);

  // BPR + reg losses
  xs_bpr_reg<<<blk((long long)BSZ * 32, TB), TB, 0, stream>>>(
      ue, pe, ne, user_table, item_table, user, positive, negative, out);

  // InfoNCE (users: ucl vs ue; items: icl vs pe)
  const float* v1s[2] = {ucl, icl};
  const float* v2s[2] = {ue, pe};
  for (int r = 0; r < 2; ++r) {
    xs_norms_pos<<<blk((long long)BSZ * 32, TB), TB, 0, stream>>>(
        v1s[r], v2s[r], n1, n2, poss);
    xs_pack<<<blk(256 * 2 * 32 * 16, TB), TB, 0, stream>>>(v1s[r], n1, fragA, 0);
    xs_pack<<<blk(256 * 2 * 32 * 16, TB), TB, 0, stream>>>(v2s[r], n2, fragB, 1);
    xs_gemm_lse<<<blk(256 * 32, TB), TB, 0, stream>>>(fragA, fragB, rowsum);
    xs_ssl_fin<<<blk(BSZ, TB), TB, 0, stream>>>(rowsum, poss, out);
  }
}
